// ElmanRnn_49701361549663
// MI455X (gfx1250) — matrix operand order repaired and checked by reading, not compiled, not run on hardware
//
#include <hip/hip_runtime.h>
#include <hip/hip_bf16.h>
#include <math.h>

typedef __attribute__((ext_vector_type(16))) _Float16 v16h;
typedef __attribute__((ext_vector_type(8)))  _Float16 v8h;
typedef __attribute__((ext_vector_type(8)))  float    v8f;

#define RNN_B 256
#define RNN_S 2048
#define RNN_I 128
#define RNN_H 256
#define BTILE 16          // batch rows per workgroup
#define NWAVES 16         // one 16x16 N-tile per wave -> covers H=256
#define THREADS (NWAVES * 32)

// LDS layout (in _Float16 elements); h and x tiles double-buffered
#define OFF_WHH 0                   // 256 x 256      = 65536
#define OFF_WIH 65536               // 256 x 128      = 32768
#define OFF_H   98304               // 2 * (16 x 256) = 8192
#define OFF_X   106496              // 2 * (16 x 128) = 4096
#define LDS_HALFS 110592
#define LDS_BYTES (LDS_HALFS * 2)   // 221184 B < 320KB WGP LDS

// Barrier that only orders LDS (dscnt), NOT the fire-and-forget global
// out-stores (storecnt). Mnemonics match compiler-emitted gfx1250 asm.
__device__ __forceinline__ void wg_barrier() {
    asm volatile("s_wait_dscnt 0x0\n\t"
                 "s_barrier_signal -1\n\t"
                 "s_barrier_wait -1" ::: "memory");
}

__device__ __forceinline__ float fast_tanh(float x) {
#if __has_builtin(__builtin_amdgcn_tanhf)
    return __builtin_amdgcn_tanhf(x);            // native v_tanh_f32
#else
    return 1.0f - 2.0f / (1.0f + __expf(2.0f * x));  // branch-free, saturating
#endif
}

// A-matrix fragment (16x32 f16): lane L holds row M=L%16.
// lane-half selects K 0-15 vs 16-31, contiguous (matches B)
// -> two contiguous 16B LDS loads.
__device__ __forceinline__ v16h ldA(const _Float16* base, int row, int rowStride, int k0) {
    const _Float16* p = base + row * rowStride + k0;
    v8h lo = *(const v8h*)(p);
    v8h hi = *(const v8h*)(p + 8);
    v16h r;
#pragma unroll
    for (int i = 0; i < 8; ++i) { r[i] = lo[i]; r[i + 8] = hi[i]; }
    return r;
}

// B-matrix fragment (32x16 f16): lane L holds col N=L%16; lane-half selects
// K 0-15 vs 16-31, contiguous -> one 32B LDS load.
__device__ __forceinline__ v16h ldB(const _Float16* base, int row, int rowStride, int k0) {
    return *(const v16h*)(base + row * rowStride + k0);
}

__global__ void __launch_bounds__(THREADS)
elman_rnn_persistent(const float* __restrict__ x_in,   // (B,S,I)
                     const float* __restrict__ W_ih,   // (H,I)
                     const float* __restrict__ W_hh,   // (H,H)
                     const float* __restrict__ b_ih,   // (H)
                     const float* __restrict__ b_hh,   // (H)
                     float* __restrict__ out)          // (B,S,H)
{
    extern __shared__ _Float16 smem[];
    _Float16* sWhh = smem + OFF_WHH;
    _Float16* sWih = smem + OFF_WIH;
    _Float16* sH   = smem + OFF_H;    // [2][16*256]
    _Float16* sX   = smem + OFF_X;    // [2][16*128]

    const int j     = threadIdx.x;
    const int lane  = j & 31;
    const int lhalf = lane >> 4;
    const int lmod  = lane & 15;
    const int n0    = (j >> 5) * 16;       // wave's hidden-column tile
    const int n     = n0 + lmod;           // column owned by this lane
    const int m0    = blockIdx.x * BTILE;

    // ---- one-time: weights f32 -> f16 into LDS; zero h ----
    for (int e = j; e < RNN_H * RNN_H; e += THREADS) sWhh[e] = (_Float16)W_hh[e];
    for (int e = j; e < RNN_H * RNN_I; e += THREADS) sWih[e] = (_Float16)W_ih[e];
    for (int e = j; e < BTILE * RNN_H; e += THREADS) sH[e] = (_Float16)0.0f;

    // ---- x prefetch pipeline (depth 4): one float4 per thread per step ----
    const int m_r  = j >> 5;               // 0..15 (row in batch tile)
    const int iofs = (j & 31) * 4;         // 0..124 (col in I)
    const float* xrow = x_in + (size_t)(m0 + m_r) * RNN_S * RNN_I + iofs;
    float4 xv[4];
#pragma unroll
    for (int u = 0; u < 4; ++u) xv[u] = *(const float4*)(xrow + (size_t)u * RNN_I);
    {   // stage x(t=0) into xb[0]
        _Float16* d = sX + m_r * RNN_I + iofs;
        d[0] = (_Float16)xv[0].x; d[1] = (_Float16)xv[0].y;
        d[2] = (_Float16)xv[0].z; d[3] = (_Float16)xv[0].w;
    }
    __syncthreads();

    // ---- hoist B-operand fragments (weights) into registers ----
    v16h wih_f[4];
#pragma unroll
    for (int kb = 0; kb < 4; ++kb)
        wih_f[kb] = ldB(sWih, n, RNN_I, kb * 32 + lhalf * 16);
    v16h whh_f[8];
#pragma unroll
    for (int kb = 0; kb < 8; ++kb)
        whh_f[kb] = ldB(sWhh, n, RNN_H, kb * 32 + lhalf * 16);

    const float bias = b_ih[n] + b_hh[n];
    const size_t SH = (size_t)RNN_S * RNN_H;

    for (int t = 0; t < RNN_S; t += 4) {
#pragma unroll
        for (int u = 0; u < 4; ++u) {
            const int tt = t + u;
            const int pr = u & 1;          // buffers read this step
            const int pw = pr ^ 1;         // buffers written this step

            // issue global load for x(tt+4); consumed 3 steps from now
            int tl = tt + 4; if (tl > RNN_S - 1) tl = RNN_S - 1;
            xv[u] = *(const float4*)(xrow + (size_t)tl * RNN_I);

            // stage x(tt+1) from registers into xb[pw]
            {
                const float4 s = xv[(u + 1) & 3];
                _Float16* d = sX + pw * (BTILE * RNN_I) + m_r * RNN_I + iofs;
                d[0] = (_Float16)s.x; d[1] = (_Float16)s.y;
                d[2] = (_Float16)s.z; d[3] = (_Float16)s.w;
            }

            // ---- two independent WMMA chains: input proj + recurrence ----
            v8f acc0, acc1;
#pragma unroll
            for (int r = 0; r < 8; ++r) { acc0[r] = bias; acc1[r] = 0.0f; }

            const _Float16* xb = sX + pr * (BTILE * RNN_I);
            const _Float16* hb = sH + pr * (BTILE * RNN_H);
#pragma unroll
            for (int kb = 0; kb < 4; ++kb) {
                v16h a = ldA(xb, lmod, RNN_I, kb * 32 + lhalf * 16);
                acc0 = __builtin_amdgcn_wmma_f32_16x16x32_f16(
                    false, a, false, wih_f[kb], (short)0, acc0, false, false);
            }
#pragma unroll
            for (int kb = 0; kb < 8; ++kb) {
                v16h a = ldA(hb, lmod, RNN_H, kb * 32 + lhalf * 16);
                acc1 = __builtin_amdgcn_wmma_f32_16x16x32_f16(
                    false, a, false, whh_f[kb], (short)0, acc1, false, false);
            }

            // ---- tanh, h -> hb[pw] (f16), out -> HBM (f32, no wait) ----
            _Float16* hw = sH + pw * (BTILE * RNN_H);
#pragma unroll
            for (int r = 0; r < 8; ++r) {
                const int mr = r + 8 * lhalf;      // C/D: M = r + 8*(lane/16)
                const float v = fast_tanh(acc0[r] + acc1[r]);
                out[(size_t)(m0 + mr) * SH + (size_t)tt * RNN_H + n] = v;
                hw[mr * RNN_H + n] = (_Float16)v;
            }
            wg_barrier();   // orders LDS only; out-stores stay in flight
        }
    }
}

extern "C" void kernel_launch(void* const* d_in, const int* in_sizes, int n_in,
                              void* d_out, int out_size, void* d_ws, size_t ws_size,
                              hipStream_t stream) {
    const float* x_in = (const float*)d_in[0];
    const float* W_ih = (const float*)d_in[1];
    const float* W_hh = (const float*)d_in[2];
    const float* b_ih = (const float*)d_in[3];
    const float* b_hh = (const float*)d_in[4];
    float* out = (float*)d_out;
    (void)in_sizes; (void)n_in; (void)out_size; (void)d_ws; (void)ws_size;

    elman_rnn_persistent<<<dim3(RNN_B / BTILE), dim3(THREADS), LDS_BYTES, stream>>>(
        x_in, W_ih, W_hh, b_ih, b_hh, out);
}
